// MinkGlobalEnc_13288628814175
// MI455X (gfx1250) — compile-verified, matrix operand order checked by reading, not run on hardware
//
#include <hip/hip_runtime.h>

typedef __attribute__((ext_vector_type(16))) _Float16 v16h;
typedef __attribute__((ext_vector_type(8)))  _Float16 v8h;
typedef __attribute__((ext_vector_type(8)))  float    v8f;
typedef __attribute__((ext_vector_type(4)))  float    v4f;

#define EPS_BN 1e-5f

struct Dims { int D, H, W; };

// ---------------- small utility kernels ----------------

__global__ void k_zero(float* __restrict__ p, int n) {
  int i = blockIdx.x * blockDim.x + threadIdx.x;
  if (i < n) p[i] = 0.0f;
}

__global__ void k_mask0(const float* __restrict__ occ, float* __restrict__ m,
                        float* __restrict__ cnt, int n) {
  int i = blockIdx.x * blockDim.x + threadIdx.x;
  if (i >= n) return;
  float a = occ[i] > 0.9f ? 1.0f : 0.0f;
  m[i] = a;
  if (a != 0.0f) atomicAdd(cnt, 1.0f);
}

__global__ void k_maskdown(const float* __restrict__ mi, float* __restrict__ mo,
                           float* __restrict__ cnt, int Do, int Ho, int Wo) {
  int n = Do * Ho * Wo;
  int i = blockIdx.x * blockDim.x + threadIdx.x;
  if (i >= n) return;
  int z = i / (Ho * Wo), r = i % (Ho * Wo), y = r / Wo, x = r % Wo;
  int Hi = Ho * 2, Wi = Wo * 2;
  float a = 0.0f;
  for (int dz = 0; dz < 2; ++dz)
    for (int dy = 0; dy < 2; ++dy)
      for (int dx = 0; dx < 2; ++dx)
        a = fmaxf(a, mi[((size_t)(2 * z + dz) * Hi + (2 * y + dy)) * Wi + (2 * x + dx)]);
  mo[i] = a;
  if (a != 0.0f) atomicAdd(cnt, 1.0f);
}

// x_feats [3, nvox] fp32 (masked) -> NDHWC f16 with cp=32 (channels 3..31 zero)
__global__ void k_prep(const float* __restrict__ xf, const float* __restrict__ m,
                       _Float16* __restrict__ out, int nvox) {
  int i = blockIdx.x * blockDim.x + threadIdx.x;
  if (i >= nvox * 32) return;
  int v = i >> 5, c = i & 31;
  float val = (c < 3) ? xf[(size_t)c * nvox + v] * m[v] : 0.0f;
  out[i] = (_Float16)val;
}

// Repack fp32 [cout][cin][ntaps] weights into WMMA A-fragment order:
// dst[((cot*kchunks + kc)*32 + lane)*16 + j], K index = tap*cin_pad + ci.
// Lane<16 holds K {0..7,16..23} of the 32-chunk; lane>=16 holds {8..15,24..31}.
__global__ void k_repack(const float* __restrict__ w, _Float16* __restrict__ dst,
                         int cout, int cin, int cin_pad, int ntaps, int kchunks, int cotiles) {
  int total = cotiles * kchunks * 512;
  int i = blockIdx.x * blockDim.x + threadIdx.x;
  if (i >= total) return;
  int j = i & 15, lane = (i >> 4) & 31, t = i >> 9;
  int kc = t % kchunks, cot = t / kchunks;
  int klocal = (lane < 16) ? ((j < 8) ? j : j + 8) : ((j < 8) ? j + 8 : j + 16);
  int k = kc * 32 + klocal;
  int tap = k / cin_pad, ci = k % cin_pad;
  int co = cot * 16 + (lane & 15);
  float val = 0.0f;
  if (co < cout && ci < cin && tap < ntaps)
    val = w[((size_t)co * cin + ci) * ntaps + tap];
  dst[i] = (_Float16)val;
}

// ---------------- implicit-GEMM conv via WMMA ----------------
// One wave computes a 32cout x 16voxel f32 tile (two 16x16 accumulators sharing
// one B fragment). Tap-outer loop: coordinates/bounds once per tap; inner loop
// over CC channel chunks is pure pointer-offset b128 loads + 2 v_wmma.
template <int KS, int CC>
__global__ __launch_bounds__(256) void k_conv_wmma(
    const _Float16* __restrict__ xin, const _Float16* __restrict__ wfrag,
    float* __restrict__ yraw, const float* __restrict__ masko, float* __restrict__ sums,
    int Din, int Hin, int Win, int Do, int Ho, int Wo,
    int stride, int pad, int cotiles2, int cp_out) {
  constexpr int NTAPS = KS * KS * KS;
  constexpr int CIN = CC * 32;
  constexpr int KCH = NTAPS * CC;
  int nvox = Do * Ho * Wo;
  int vtiles = (nvox + 15) >> 4;
  int ntiles = vtiles * cotiles2;
  int tile = blockIdx.x * (blockDim.x >> 5) + (threadIdx.x >> 5);
  if (tile >= ntiles) return;                 // wave-uniform exit (EXEC all-1 at WMMA)
  int lane = threadIdx.x & 31;
  int cot2 = tile % cotiles2;                 // co tiles fastest: waves in a block share B
  int vt = tile / cotiles2;
  int nn = lane & 15;
  int hi = lane >> 4;
  int v = vt * 16 + nn;
  bool vv = v < nvox;
  int vq = vv ? v : 0;
  int z = vq / (Ho * Wo);
  int rr = vq % (Ho * Wo);
  int y = rr / Wo, x = rr % Wo;
  int bz = z * stride - pad, by = y * stride - pad, bx = x * stride - pad;
  int cbase = cot2 * 32 + hi * 8;             // D layout: vgpr r -> co = r (+8 for hi half)

  float mval = vv ? masko[v] : 0.0f;          // conv(x)*mask; skip fully-inactive tiles
  if (!__any(mval != 0.0f)) {
    if (vv) {
      float* yp = yraw + (size_t)v * cp_out + cbase;
      v4f zz = {};
      *(v4f*)yp = zz; *(v4f*)(yp + 4) = zz;
      *(v4f*)(yp + 16) = zz; *(v4f*)(yp + 20) = zz;
    }
    return;
  }

  v8f acc0 = {}, acc1 = {};
  const v16h* wf0 = (const v16h*)wfrag + (size_t)(cot2 * 2) * KCH * 32 + lane;
  const v16h* wf1 = wf0 + (size_t)KCH * 32;
  for (int tap = 0; tap < NTAPS; ++tap) {
    int dz = tap / (KS * KS), dy = (tap / KS) % KS, dx = tap % KS;
    int iz = bz + dz, iy = by + dy, ix = bx + dx;
    bool inb = vv && (unsigned)iz < (unsigned)Din && (unsigned)iy < (unsigned)Hin &&
               (unsigned)ix < (unsigned)Win;
    const _Float16* px = xin + (((size_t)iz * Hin + iy) * Win + ix) * CIN + hi * 8;
#pragma unroll
    for (int cc = 0; cc < CC; ++cc) {
      int kc = tap * CC + cc;
      v16h a0 = wf0[(size_t)kc * 32];         // pre-swizzled A fragments, 32B aligned
      v16h a1 = wf1[(size_t)kc * 32];
      v16h b;
#pragma unroll
      for (int half = 0; half < 2; ++half) {  // 8 consecutive channels per run
        v8h d = {};
        if (inb) d = *(const v8h*)(px + cc * 32 + half * 16);
#pragma unroll
        for (int jj = 0; jj < 8; ++jj) b[half * 8 + jj] = d[jj];
      }
      acc0 = __builtin_amdgcn_wmma_f32_16x16x32_f16(false, a0, false, b, (short)0, acc0,
                                                    false, false);
      acc1 = __builtin_amdgcn_wmma_f32_16x16x32_f16(false, a1, false, b, (short)0, acc1,
                                                    false, false);
    }
  }

#pragma unroll
  for (int r2 = 0; r2 < 8; ++r2) { acc0[r2] *= mval; acc1[r2] *= mval; }

  if (vv) {
    float* yp = yraw + (size_t)v * cp_out + cbase;
    v4f a = {acc0[0], acc0[1], acc0[2], acc0[3]};
    v4f b4 = {acc0[4], acc0[5], acc0[6], acc0[7]};
    v4f c4 = {acc1[0], acc1[1], acc1[2], acc1[3]};
    v4f d4 = {acc1[4], acc1[5], acc1[6], acc1[7]};
    *(v4f*)yp = a; *(v4f*)(yp + 4) = b4;
    *(v4f*)(yp + 16) = c4; *(v4f*)(yp + 20) = d4;
  }

  // per-channel sum/sumsq over the 16 voxels of this tile (lane butterfly, width 16)
  float s0[8], q0[8], s1[8], q1[8];
#pragma unroll
  for (int r2 = 0; r2 < 8; ++r2) {
    s0[r2] = acc0[r2]; q0[r2] = acc0[r2] * acc0[r2];
    s1[r2] = acc1[r2]; q1[r2] = acc1[r2] * acc1[r2];
  }
#pragma unroll
  for (int m2 = 8; m2 >= 1; m2 >>= 1) {
#pragma unroll
    for (int r2 = 0; r2 < 8; ++r2) {
      s0[r2] += __shfl_xor(s0[r2], m2, 16);
      q0[r2] += __shfl_xor(q0[r2], m2, 16);
      s1[r2] += __shfl_xor(s1[r2], m2, 16);
      q1[r2] += __shfl_xor(q1[r2], m2, 16);
    }
  }
  if (nn == 0) {
#pragma unroll
    for (int r2 = 0; r2 < 8; ++r2) {
      atomicAdd(&sums[cbase + r2], s0[r2]);
      atomicAdd(&sums[cp_out + cbase + r2], q0[r2]);
      atomicAdd(&sums[cbase + 16 + r2], s1[r2]);
      atomicAdd(&sums[cp_out + cbase + 16 + r2], q1[r2]);
    }
  }
}

// BN(normalize over active count) + optional residual + optional ReLU,
// re-mask, emit f16 NDHWC for next layer; optionally fp32 NCDHW (final output).
__global__ void k_finalize(const float* __restrict__ yraw, const float* __restrict__ sums,
                           const float* __restrict__ cnt, const float* __restrict__ g,
                           const float* __restrict__ bta, const float* __restrict__ mask,
                           const _Float16* __restrict__ resid, _Float16* __restrict__ out16,
                           float* __restrict__ outf32, int nvox, int cp, int cout, int relu) {
  int i = blockIdx.x * blockDim.x + threadIdx.x;
  if (i >= nvox * cp) return;
  int v = i / cp, c = i % cp;
  float val = 0.0f;
  if (c < cout) {
    float n = fmaxf(cnt[0], 1.0f);
    float mean = sums[c] / n;
    float var = sums[cp + c] / n - mean * mean;
    float yv = yraw[i];
    val = (yv - mean) * rsqrtf(var + EPS_BN) * g[c] + bta[c];
    val *= mask[v];
    if (resid) val += (float)resid[i];
    if (relu) val = fmaxf(val, 0.0f);
    if (outf32) outf32[(size_t)c * nvox + v] = val;
  }
  out16[i] = (_Float16)val;
}

// ---------------- host orchestration ----------------

extern "C" void kernel_launch(void* const* d_in, const int* in_sizes, int n_in,
                              void* d_out, int out_size, void* d_ws, size_t ws_size,
                              hipStream_t stream) {
  (void)in_sizes; (void)n_in; (void)out_size; (void)ws_size;
#define P(i) ((const float*)d_in[i])
  const float* xf = P(0);
  const float* occ = P(1);

  const Dims L0{64, 64, 64}, L1{32, 32, 32}, L2{16, 16, 16}, L3{8, 8, 8}, L4{4, 4, 4};
  const int NV0 = 262144, NV1 = 32768, NV2 = 4096, NV3 = 512, NV4 = 64;

  char* ws = (char*)d_ws;
  size_t cur = 0;
  auto alloc = [&](size_t b) { size_t o = cur; cur += (b + 255) & ~(size_t)255; return o; };
  float* m0 = (float*)(ws + alloc((size_t)NV0 * 4));
  float* m1 = (float*)(ws + alloc((size_t)NV1 * 4));
  float* m2 = (float*)(ws + alloc((size_t)NV2 * 4));
  float* m3 = (float*)(ws + alloc((size_t)NV3 * 4));
  float* m4 = (float*)(ws + alloc((size_t)NV4 * 4));
  float* cnts = (float*)(ws + alloc(8 * 4));
  float* sums = (float*)(ws + alloc(512 * 4));
  float* yraw = (float*)(ws + alloc((size_t)NV0 * 32 * 4));
  _Float16* bufA = (_Float16*)(ws + alloc((size_t)NV0 * 32 * 2));
  _Float16* bufB = (_Float16*)(ws + alloc((size_t)NV0 * 32 * 2));
  _Float16* bufC = (_Float16*)(ws + alloc((size_t)NV0 * 32 * 2));
  _Float16* wfa = (_Float16*)(ws + alloc((size_t)8600000 * 2));  // fragment arena (~17MB used)

  auto cdiv = [](int a, int b) { return (a + b - 1) / b; };

  // masks + active-voxel counts per level
  k_zero<<<1, 32, 0, stream>>>(cnts, 8);
  k_mask0<<<cdiv(NV0, 256), 256, 0, stream>>>(occ, m0, &cnts[0], NV0);
  k_maskdown<<<cdiv(NV1, 256), 256, 0, stream>>>(m0, m1, &cnts[1], 32, 32, 32);
  k_maskdown<<<cdiv(NV2, 256), 256, 0, stream>>>(m1, m2, &cnts[2], 16, 16, 16);
  k_maskdown<<<cdiv(NV3, 256), 256, 0, stream>>>(m2, m3, &cnts[3], 8, 8, 8);
  k_maskdown<<<cdiv(NV4, 256), 256, 0, stream>>>(m3, m4, &cnts[4], 4, 4, 4);
  k_prep<<<cdiv(NV0 * 32, 256), 256, 0, stream>>>(xf, m0, bufC, NV0);

  size_t wcur = 0;  // deterministic weight-fragment packing order every call
  auto conv = [&](const float* w, const _Float16* xin, Dims din, Dims dout,
                  int cin, int cout, int ks, int stride, int pad, const float* maskO) {
    int cin_pad = (cin < 32) ? 32 : cin;
    int ntaps = ks * ks * ks;
    int cc = cin_pad / 32;
    int kchunks = ntaps * cc;
    int cotiles = cout / 16;
    int cotiles2 = cout / 32;
    int cp_out = cout;
    _Float16* frag = wfa + wcur;
    wcur += (size_t)cotiles * kchunks * 512;
    int rtot = cotiles * kchunks * 512;
    k_repack<<<cdiv(rtot, 256), 256, 0, stream>>>(w, frag, cout, cin, cin_pad, ntaps,
                                                  kchunks, cotiles);
    k_zero<<<2, 256, 0, stream>>>(sums, 2 * cp_out);
    int nvox = dout.D * dout.H * dout.W;
    int ntiles = cdiv(nvox, 16) * cotiles2;
    int blocks = cdiv(ntiles, 8);
#define LC(KS, CCV)                                                                   \
  k_conv_wmma<KS, CCV><<<blocks, 256, 0, stream>>>(                                   \
      xin, frag, yraw, maskO, sums, din.D, din.H, din.W, dout.D, dout.H, dout.W,      \
      stride, pad, cotiles2, cp_out)
    if (ks == 3) {
      if (cc == 1) LC(3, 1); else if (cc == 2) LC(3, 2);
      else if (cc == 4) LC(3, 4); else LC(3, 8);
    } else if (ks == 2) {
      if (cc == 1) LC(2, 1); else if (cc == 2) LC(2, 2); else LC(2, 4);
    } else {
      if (cc == 1) LC(1, 1); else if (cc == 2) LC(1, 2); else LC(1, 4);
    }
#undef LC
  };
  auto fin = [&](const float* g, const float* bb, const float* maskO, float* cnt,
                 const _Float16* resid, _Float16* out16, float* outf32,
                 int nvox, int cp, int cout, int relu) {
    int tot = nvox * cp;
    k_finalize<<<cdiv(tot, 256), 256, 0, stream>>>(yraw, sums, cnt, g, bb, maskO, resid,
                                                   out16, outf32, nvox, cp, cout, relu);
  };

  // ---- stem (level 0) ----
  conv(P(75), bufC, L0, L0, 3, 32, 3, 1, 1, m0);
  fin(P(73), P(71), m0, &cnts[0], nullptr, bufA, nullptr, NV0, 32, 32, 1);
  conv(P(76), bufA, L0, L0, 32, 32, 3, 1, 1, m0);
  fin(P(74), P(72), m0, &cnts[0], nullptr, bufB, nullptr, NV0, 32, 32, 1);

  // ---- stage1 (level 1, 32->32) ----
  conv(P(4), bufB, L0, L1, 32, 32, 2, 2, 0, m1);
  fin(P(3), P(2), m1, &cnts[1], nullptr, bufA, nullptr, NV1, 32, 32, 1);
  conv(P(9), bufA, L1, L1, 32, 32, 3, 1, 1, m1);
  fin(P(7), P(5), m1, &cnts[1], nullptr, bufB, nullptr, NV1, 32, 32, 1);
  conv(P(10), bufB, L1, L1, 32, 32, 3, 1, 1, m1);
  fin(P(8), P(6), m1, &cnts[1], bufA, bufC, nullptr, NV1, 32, 32, 1);
  conv(P(15), bufC, L1, L1, 32, 32, 3, 1, 1, m1);
  fin(P(13), P(11), m1, &cnts[1], nullptr, bufA, nullptr, NV1, 32, 32, 1);
  conv(P(16), bufA, L1, L1, 32, 32, 3, 1, 1, m1);
  fin(P(14), P(12), m1, &cnts[1], bufC, bufB, nullptr, NV1, 32, 32, 1);

  // ---- stage2 (level 2, 32->64) ----
  conv(P(19), bufB, L1, L2, 32, 32, 2, 2, 0, m2);
  fin(P(18), P(17), m2, &cnts[2], nullptr, bufA, nullptr, NV2, 32, 32, 1);
  conv(P(28), bufA, L2, L2, 32, 64, 1, 1, 0, m2);                // shortcut wd
  fin(P(25), P(22), m2, &cnts[2], nullptr, bufC, nullptr, NV2, 64, 64, 0);
  conv(P(26), bufA, L2, L2, 32, 64, 3, 1, 1, m2);
  fin(P(23), P(20), m2, &cnts[2], nullptr, bufB, nullptr, NV2, 64, 64, 1);
  conv(P(27), bufB, L2, L2, 64, 64, 3, 1, 1, m2);
  fin(P(24), P(21), m2, &cnts[2], bufC, bufA, nullptr, NV2, 64, 64, 1);
  conv(P(33), bufA, L2, L2, 64, 64, 3, 1, 1, m2);
  fin(P(31), P(29), m2, &cnts[2], nullptr, bufB, nullptr, NV2, 64, 64, 1);
  conv(P(34), bufB, L2, L2, 64, 64, 3, 1, 1, m2);
  fin(P(32), P(30), m2, &cnts[2], bufA, bufC, nullptr, NV2, 64, 64, 1);

  // ---- stage3 (level 3, 64->128) ----
  conv(P(37), bufC, L2, L3, 64, 64, 2, 2, 0, m3);
  fin(P(36), P(35), m3, &cnts[3], nullptr, bufA, nullptr, NV3, 64, 64, 1);
  conv(P(46), bufA, L3, L3, 64, 128, 1, 1, 0, m3);               // shortcut wd
  fin(P(43), P(40), m3, &cnts[3], nullptr, bufC, nullptr, NV3, 128, 128, 0);
  conv(P(44), bufA, L3, L3, 64, 128, 3, 1, 1, m3);
  fin(P(41), P(38), m3, &cnts[3], nullptr, bufB, nullptr, NV3, 128, 128, 1);
  conv(P(45), bufB, L3, L3, 128, 128, 3, 1, 1, m3);
  fin(P(42), P(39), m3, &cnts[3], bufC, bufA, nullptr, NV3, 128, 128, 1);
  conv(P(51), bufA, L3, L3, 128, 128, 3, 1, 1, m3);
  fin(P(49), P(47), m3, &cnts[3], nullptr, bufB, nullptr, NV3, 128, 128, 1);
  conv(P(52), bufB, L3, L3, 128, 128, 3, 1, 1, m3);
  fin(P(50), P(48), m3, &cnts[3], bufA, bufC, nullptr, NV3, 128, 128, 1);

  // ---- stage4 (level 4, 128->256) ----
  conv(P(55), bufC, L3, L4, 128, 128, 2, 2, 0, m4);
  fin(P(54), P(53), m4, &cnts[4], nullptr, bufA, nullptr, NV4, 128, 128, 1);
  conv(P(64), bufA, L4, L4, 128, 256, 1, 1, 0, m4);              // shortcut wd
  fin(P(61), P(58), m4, &cnts[4], nullptr, bufC, nullptr, NV4, 256, 256, 0);
  conv(P(62), bufA, L4, L4, 128, 256, 3, 1, 1, m4);
  fin(P(59), P(56), m4, &cnts[4], nullptr, bufB, nullptr, NV4, 256, 256, 1);
  conv(P(63), bufB, L4, L4, 256, 256, 3, 1, 1, m4);
  fin(P(60), P(57), m4, &cnts[4], bufC, bufA, nullptr, NV4, 256, 256, 1);
  conv(P(69), bufA, L4, L4, 256, 256, 3, 1, 1, m4);
  fin(P(67), P(65), m4, &cnts[4], nullptr, bufB, nullptr, NV4, 256, 256, 1);
  conv(P(70), bufB, L4, L4, 256, 256, 3, 1, 1, m4);
  fin(P(68), P(66), m4, &cnts[4], bufA, bufC, (float*)d_out, NV4, 256, 256, 1);
#undef P
}
